// QKNet_54219667145469
// MI455X (gfx1250) — matrix operand order repaired and checked by reading, not compiled, not run on hardware
//
#include <hip/hip_runtime.h>

// ---------------------------------------------------------------------------
// Types for CDNA5 WMMA (wave32)
// ---------------------------------------------------------------------------
typedef __attribute__((ext_vector_type(16))) __bf16 v16bf;
typedef __attribute__((ext_vector_type(8)))  __bf16 v8bf;
typedef __attribute__((ext_vector_type(8)))  float  v8f;

#define QK_ALPHA 0.9f

static __device__ __forceinline__ __bf16 f2bf(float f) {
  unsigned u = __float_as_uint(f);
  unsigned r = (u + 0x7FFFu + ((u >> 16) & 1u)) >> 16;   // round-to-nearest-even
  unsigned short h = (unsigned short)r;
  return __builtin_bit_cast(__bf16, h);
}
static __device__ __forceinline__ float bf2f(__bf16 b) {
  unsigned short h = __builtin_bit_cast(unsigned short, b);
  return __uint_as_float(((unsigned)h) << 16);
}
static __device__ __forceinline__ v16bf cat16(v8bf lo, v8bf hi) {
  return __builtin_shufflevector(lo, hi, 0,1,2,3,4,5,6,7,8,9,10,11,12,13,14,15);
}

// ---------------------------------------------------------------------------
// conv1 (1->32ch, 5x5 pad2, 28x28) + ReLU + 2x2 maxpool  -> NCHW f32 (B,32,196)
// ---------------------------------------------------------------------------
__global__ void qk_conv1_relu_pool(const float* __restrict__ x, const float* __restrict__ w,
                                   const float* __restrict__ bias, float* __restrict__ out,
                                   int total) {
  int t = blockIdx.x * blockDim.x + threadIdx.x;
  if (t >= total) return;
  int ox = t % 14; int tt = t / 14;
  int oy = tt % 14; tt /= 14;
  int co = tt % 32; int b = tt / 32;
  const float* wp = w + co * 25;
  const float* xp = x + (size_t)b * 784;
  float bb = bias[co];
  float m = -3.4e38f;
  #pragma unroll
  for (int py = 0; py < 2; ++py)
  #pragma unroll
  for (int px = 0; px < 2; ++px) {
    int y = oy * 2 + py, xx = ox * 2 + px;
    float acc = bb;
    #pragma unroll
    for (int ky = 0; ky < 5; ++ky) {
      int sy = y + ky - 2;
      if (sy < 0 || sy >= 28) continue;
      #pragma unroll
      for (int kx = 0; kx < 5; ++kx) {
        int sx = xx + kx - 2;
        if (sx < 0 || sx >= 28) continue;
        acc = fmaf(xp[sy * 28 + sx], wp[ky * 5 + kx], acc);
      }
    }
    m = fmaxf(m, fmaxf(acc, 0.0f));
  }
  out[((size_t)b * 32 + co) * 196 + oy * 14 + ox] = m;
}

// ---------------------------------------------------------------------------
// Row L2-normalize: h (rows x D) f32 -> xn (rows x Dp) bf16, zero-padded
// one wave per row
// ---------------------------------------------------------------------------
__global__ void qk_knn_normalize(const float* __restrict__ h, __bf16* __restrict__ xn,
                                 int rows, int D, int Dp) {
  int wid  = (blockIdx.x * blockDim.x + threadIdx.x) >> 5;
  int lane = threadIdx.x & 31;
  if (wid >= rows) return;
  const float* hp = h + (size_t)wid * D;
  float ss = 0.f;
  for (int d = lane; d < D; d += 32) { float v = hp[d]; ss = fmaf(v, v, ss); }
  #pragma unroll
  for (int off = 16; off > 0; off >>= 1) ss += __shfl_xor(ss, off);
  float inv = 1.0f / fmaxf(sqrtf(ss), 1e-12f);
  __bf16* op = xn + (size_t)wid * Dp;
  for (int d = lane; d < Dp; d += 32)
    op[d] = f2bf(d < D ? hp[d] * inv : 0.0f);
}

// centers f32 (C,K,D) -> bf16 (C,Kp,Dp) zero-padded
__global__ void qk_cast_centers(const float* __restrict__ c, __bf16* __restrict__ out,
                                int C, int K, int Kp, int D, int Dp) {
  long t = (long)blockIdx.x * blockDim.x + threadIdx.x;
  long total = (long)C * Kp * Dp;
  if (t >= total) return;
  int dp = (int)(t % Dp); long tt = t / Dp;
  int k  = (int)(tt % Kp); int cc = (int)(tt / Kp);
  float v = (k < K && dp < D) ? c[((size_t)cc * K + k) * D + dp] : 0.0f;
  out[t] = f2bf(v);
}

// ---------------------------------------------------------------------------
// Fused distance GEMM + argmax (== argmin of 1-dot) + counts.
// A = xn, row (b*C+c), K=DP (compile-time).  B = cent (C,Kp,DP).
// One wave per (c, 16-row tile); the whole 16xDP A panel is register-resident
// and reused across all 13 N-tiles.
// ---------------------------------------------------------------------------
template <int DP>
__global__ void qk_knn_dist_argmax(const __bf16* __restrict__ xn, const __bf16* __restrict__ cent,
                                   int* __restrict__ idx, float* __restrict__ counts,
                                   int Bsz, int C, int K, int Kp) {
  constexpr int NK = DP / 32;
  int gw   = (blockIdx.x * blockDim.x + threadIdx.x) >> 5;
  int lane = threadIdx.x & 31;
  int mtiles = Bsz >> 4;
  if (gw >= C * mtiles) return;
  int c = gw / mtiles, mt = gw % mtiles;
  int n = lane & 15, half = lane >> 4;
  const __bf16* Abase = xn + ((size_t)(mt * 16 + n) * C + c) * DP;
  const __bf16* Bbase = cent + (size_t)c * Kp * DP;
  v8bf alo[NK], ahi[NK];
  #pragma unroll
  for (int ks = 0; ks < NK; ++ks) {
    alo[ks] = *(const v8bf*)(Abase + ks * 32 + half * 8);
    ahi[ks] = *(const v8bf*)(Abase + ks * 32 + 16 + half * 8);
  }
  int ntiles = Kp >> 4;
  float bestv[8]; int besti[8];
  #pragma unroll
  for (int r = 0; r < 8; ++r) { bestv[r] = -3.4e38f; besti[r] = 0; }
  for (int nt = 0; nt < ntiles; ++nt) {
    v8f acc = {};
    const __bf16* Brow = Bbase + (size_t)(nt * 16 + n) * DP;
    #pragma unroll
    for (int ks = 0; ks < NK; ++ks) {
      v8bf blo = *(const v8bf*)(Brow + ks * 32 + half * 8);
      v8bf bhi = *(const v8bf*)(Brow + ks * 32 + 16 + half * 8);
      acc = __builtin_amdgcn_wmma_f32_16x16x32_bf16(false, cat16(alo[ks], ahi[ks]),
                                                    false, cat16(blo, bhi),
                                                    (short)0, acc, false, false);
    }
    int kcol = nt * 16 + n;
    if (kcol < K) {
      #pragma unroll
      for (int r = 0; r < 8; ++r) {
        float v = acc[r];
        if (v > bestv[r]) { bestv[r] = v; besti[r] = kcol; }
      }
    }
  }
  // reduce max-dot (min-dist) across the 16 N-lanes of each half
  #pragma unroll
  for (int off = 1; off < 16; off <<= 1) {
    #pragma unroll
    for (int r = 0; r < 8; ++r) {
      float ov = __shfl_xor(bestv[r], off);
      int   oi = __shfl_xor(besti[r], off);
      if (ov > bestv[r] || (ov == bestv[r] && oi < besti[r])) { bestv[r] = ov; besti[r] = oi; }
    }
  }
  if (n == 0) {
    #pragma unroll
    for (int r = 0; r < 8; ++r) {
      int row = mt * 16 + half * 8 + r;
      idx[(size_t)c * Bsz + row] = besti[r];
      atomicAdd(&counts[c * K + besti[r]], 1.0f);
    }
  }
}

__global__ void qk_knn_accum_sums(const __bf16* __restrict__ xn, const int* __restrict__ idx,
                                  float* __restrict__ sums, int Bsz, int C, int K, int D, int Dp,
                                  long total) {
  long t = (long)blockIdx.x * blockDim.x + threadIdx.x;
  if (t >= total) return;
  int d = (int)(t % D); long tt = t / D;
  int c = (int)(tt % C); int b = (int)(tt / C);
  int k = idx[(size_t)c * Bsz + b];
  float v = bf2f(xn[((size_t)b * C + c) * Dp + d]);
  atomicAdd(&sums[((size_t)c * K + k) * D + d], v);
}

// one wave per (c,k): mean -> EMA/first -> normalize -> select by count
__global__ void qk_knn_update(const float* __restrict__ cent, const int* __restrict__ table,
                              const float* __restrict__ counts, const float* __restrict__ sums,
                              float* __restrict__ newc, int C, int K, int D) {
  int wid  = (blockIdx.x * blockDim.x + threadIdx.x) >> 5;
  int lane = threadIdx.x & 31;
  if (wid >= C * K) return;
  float cnt = counts[wid];
  int   tab = table[wid];
  const float* sp = sums + (size_t)wid * D;
  const float* cp = cent + (size_t)wid * D;
  float denom = fmaxf(cnt, 1.0f);
  float ss = 0.f;
  for (int d = lane; d < D; d += 32) {
    float mean = sp[d] / denom;
    float u = (tab == 0) ? mean : (QK_ALPHA * cp[d] + (1.0f - QK_ALPHA) * mean);
    ss = fmaf(u, u, ss);
  }
  #pragma unroll
  for (int off = 16; off > 0; off >>= 1) ss += __shfl_xor(ss, off);
  float rn = rsqrtf(ss);
  float* op = newc + (size_t)wid * D;
  for (int d = lane; d < D; d += 32) {
    float mean = sp[d] / denom;
    float u = (tab == 0) ? mean : (QK_ALPHA * cp[d] + (1.0f - QK_ALPHA) * mean);
    op[d] = (cnt > 0.0f) ? u * rn : cp[d];
  }
}

// gather res = newc[c, idx[b,c], :]; mode 0: NHWC bf16 (b,d,c); mode 1: NCHW-flat bf16
__global__ void qk_knn_gather(const float* __restrict__ newc, const int* __restrict__ idx,
                              __bf16* __restrict__ out, int Bsz, int C, int K, int D, int mode,
                              long total) {
  long t = (long)blockIdx.x * blockDim.x + threadIdx.x;
  if (t >= total) return;
  int d = (int)(t % D); long tt = t / D;
  int c = (int)(tt % C); int b = (int)(tt / C);
  int k = idx[(size_t)c * Bsz + b];
  float v = newc[((size_t)c * K + k) * D + d];
  if (mode == 0) out[((size_t)b * D + d) * C + c] = f2bf(v);
  else           out[(size_t)b * C * D + (size_t)c * D + d] = f2bf(v);
}

// OIHW f32 (O,32,5,5) -> (25, O, 32) bf16
__global__ void qk_pack_conv_w(const float* __restrict__ w, __bf16* __restrict__ out,
                               int O, int total) {
  int t = blockIdx.x * blockDim.x + threadIdx.x;
  if (t >= total) return;
  int i = t % 32; int tt = t / 32;
  int o = tt % O; int tap = tt / O;
  out[t] = f2bf(w[((size_t)o * 32 + i) * 25 + tap]);
}

// ---------------------------------------------------------------------------
// 5x5 pad-2 conv as 25 shifted K=32 WMMA GEMMs. in: NHWC bf16 (B,14,14,32)
// wpack: (25, Cout, 32) bf16. out: NCHW f32 (B,Cout,196), ReLU+bias fused.
// one wave per 16x16 output tile.
// ---------------------------------------------------------------------------
__global__ void qk_conv_taps(const __bf16* __restrict__ in, const __bf16* __restrict__ wpack,
                             const float* __restrict__ bias, float* __restrict__ out,
                             int Bsz, int Cout) {
  int gw   = (blockIdx.x * blockDim.x + threadIdx.x) >> 5;
  int lane = threadIdx.x & 31;
  int mtiles = (Bsz * 196) >> 4;
  int ntiles = Cout >> 4;
  if (gw >= mtiles * ntiles) return;
  int mt = gw / ntiles, nt = gw % ntiles;
  int n = lane & 15, half = lane >> 4;
  int m = mt * 16 + n;
  int b = m / 196, p = m % 196, y = p / 14, x = p % 14;
  v8f acc = {};
  #pragma unroll
  for (int dy = 0; dy < 5; ++dy) {
    int sy = y + dy - 2;
    #pragma unroll
    for (int dx = 0; dx < 5; ++dx) {
      int sx = x + dx - 2;
      bool valid = (sy >= 0) & (sy < 14) & (sx >= 0) & (sx < 14);
      v8bf alo = {}, ahi = {};
      if (valid) {
        const __bf16* ap = in + ((size_t)b * 196 + sy * 14 + sx) * 32;
        alo = *(const v8bf*)(ap + half * 8);
        ahi = *(const v8bf*)(ap + 16 + half * 8);
      }
      const __bf16* wp = wpack + ((size_t)(dy * 5 + dx) * Cout + nt * 16 + n) * 32;
      v8bf blo = *(const v8bf*)(wp + half * 8);
      v8bf bhi = *(const v8bf*)(wp + 16 + half * 8);
      acc = __builtin_amdgcn_wmma_f32_16x16x32_bf16(false, cat16(alo, ahi),
                                                    false, cat16(blo, bhi),
                                                    (short)0, acc, false, false);
    }
  }
  int co = nt * 16 + n;
  float bv = bias[co];
  #pragma unroll
  for (int r = 0; r < 8; ++r) {
    int mr = mt * 16 + half * 8 + r;
    int bb = mr / 196, pp = mr % 196;
    out[((size_t)bb * Cout + co) * 196 + pp] = fmaxf(acc[r] + bv, 0.0f);
  }
}

// 2x2 maxpool (B,64,14,14) -> (B,64,7,7)
__global__ void qk_maxpool2(const float* __restrict__ in, float* __restrict__ out, int total) {
  int t = blockIdx.x * blockDim.x + threadIdx.x;
  if (t >= total) return;
  int ox = t % 7; int tt = t / 7;
  int oy = tt % 7; int cb = tt / 7;
  const float* p = in + (size_t)cb * 196;
  int y = oy * 2, x = ox * 2;
  float m = fmaxf(fmaxf(p[y * 14 + x], p[y * 14 + x + 1]),
                  fmaxf(p[(y + 1) * 14 + x], p[(y + 1) * 14 + x + 1]));
  out[(size_t)cb * 49 + oy * 7 + ox] = m;
}

__global__ void qk_cast_f32_bf16(const float* __restrict__ src, __bf16* __restrict__ dst,
                                 long total) {
  long t = (long)blockIdx.x * blockDim.x + threadIdx.x;
  if (t >= total) return;
  dst[t] = f2bf(src[t]);
}

// ---------------------------------------------------------------------------
// C = A(MxKp) * B(NpadxKp)^T (+bias, relu, optional bf16 mirror).
// Block = 4 waves sharing one N-tile (4 consecutive M-tiles). The 16x64 bf16
// B chunk is staged into LDS with CDNA5 async-to-LDS DMA, double-buffered:
// every thread issues one global_load_async_to_lds_b128 (16 B/lane), waves
// drain their ASYNCcnt, and a workgroup barrier publishes the chunk.
// Requires: M % 64 == 0, Kp % 64 == 0.
// ---------------------------------------------------------------------------
__global__ void qk_gemm_nt(const __bf16* __restrict__ A, int lda,
                           const __bf16* __restrict__ Bm, int ldb,
                           float* __restrict__ Cm, int ldc,
                           const float* __restrict__ bias, int relu,
                           __bf16* __restrict__ outbf, int ldob,
                           int M, int Nstore, int Npad, int Kp) {
  __shared__ __align__(16) __bf16 lds_b[2][16][64];   // 2 x 2 KB double buffer
  int ntiles = Npad >> 4;
  int mtg = blockIdx.x / ntiles;
  int nt  = blockIdx.x % ntiles;
  int w    = threadIdx.x >> 5;
  int lane = threadIdx.x & 31;
  int mt = mtg * 4 + w;
  int n = lane & 15, half = lane >> 4;
  const __bf16* Ap = A + (size_t)(mt * 16 + n) * lda;

  int tid  = threadIdx.x;             // 0..127 -> one 16B piece of the 2 KB chunk
  int brow = tid >> 3;                // B row within tile (0..15)
  int bcol = (tid & 7) * 8;           // element column within 64-wide chunk
  const __bf16* Bg = Bm + (size_t)(nt * 16 + brow) * ldb + bcol;

  int nchunks = Kp >> 6;
  { // stage chunk 0 into buffer 0
    unsigned lo = (unsigned)(tid * 16);
    asm volatile("global_load_async_to_lds_b128 %0, %1, off" :: "v"(lo), "v"(Bg) : "memory");
  }
  v8f acc = {};
  for (int ci = 0; ci < nchunks; ++ci) {
    int kc = ci << 6;
    if (ci + 1 < nchunks) {           // stage next chunk into the other buffer
      unsigned lo = (unsigned)((((ci + 1) & 1) * 2048) + tid * 16);
      const __bf16* gp = Bg + kc + 64;
      asm volatile("global_load_async_to_lds_b128 %0, %1, off" :: "v"(lo), "v"(gp) : "memory");
      asm volatile("s_wait_asynccnt 0x1" ::: "memory");
    } else {
      asm volatile("s_wait_asynccnt 0x0" ::: "memory");
    }
    __syncthreads();                  // chunk ci visible to all 4 waves
    const __bf16* ls = &lds_b[ci & 1][0][0];
    #pragma unroll
    for (int s = 0; s < 2; ++s) {
      int k0 = kc + s * 32;
      __builtin_prefetch(Ap + k0 + 256, 0, 1);
      v8bf alo = *(const v8bf*)(Ap + k0 + half * 8);
      v8bf ahi = *(const v8bf*)(Ap + k0 + 16 + half * 8);
      v8bf blo = *(const v8bf*)(ls + n * 64 + s * 32 + half * 8);
      v8bf bhi = *(const v8bf*)(ls + n * 64 + s * 32 + 16 + half * 8);
      acc = __builtin_amdgcn_wmma_f32_16x16x32_bf16(false, cat16(alo, ahi),
                                                    false, cat16(blo, bhi),
                                                    (short)0, acc, false, false);
    }
    __syncthreads();                  // done reading buf (ci&1) before restage
  }
  int col = nt * 16 + n;
  if (col < Nstore) {
    float bv = bias ? bias[col] : 0.0f;
    #pragma unroll
    for (int r = 0; r < 8; ++r) {
      int mr = mt * 16 + half * 8 + r;
      float v = acc[r] + bv;
      if (relu) v = fmaxf(v, 0.0f);
      Cm[(size_t)mr * ldc + col] = v;
      if (outbf) outbf[(size_t)mr * ldob + col] = f2bf(v);
    }
  }
}

// ---------------------------------------------------------------------------
// Workspace arena (bytes, 256-aligned). Peak ~105 MB with buffer reuse.
// ---------------------------------------------------------------------------
static constexpr size_t OFF_ACT    = 0;           // f32 acts: h0/h2 (25.7MB) / h3 (51.4MB)
static constexpr size_t OFF_XN     = 51380224;    // xn bf16, max 1024*32*224*2
static constexpr size_t OFF_CENT   = 66060288;    // centers bf16 padded, max 32*208*224*2
static constexpr size_t OFF_IDX    = 69042176;    // int, max 64*1024
static constexpr size_t OFF_COUNTS = 69304320;    // f32, max 64*200
static constexpr size_t OFF_SUMS   = 69355520;    // f32, max 32*200*196
static constexpr size_t OFF_NEWC   = 74373120;    // f32, max 32*200*196
static constexpr size_t OFF_QBUF   = 79390720;    // q0/q1 NHWC bf16 | h3p f32 | fcin bf16
static constexpr size_t OFF_WPACK  = 92235776;    // packed conv weights bf16, max 25*64*32
static constexpr size_t OFF_FC1W   = 92338176;    // fc1_w bf16 1024*3136
static constexpr size_t OFF_FC2W   = 98760704;    // fc2_w bf16 padded 16*1024
static constexpr size_t OFF_FC1O   = 98793472;    // fc1 out f32 1024*1024
static constexpr size_t OFF_FC1OB  = 102987776;   // fc1 out bf16 1024*1024

static inline dim3 ew_grid(long total) { return dim3((unsigned)((total + 255) / 256)); }

template <int DP>
static void run_knn(const float* h, const float* c, const int* t,
                    int C, int D, void* q_out, int mode,
                    __bf16* xn, __bf16* cent, int* idx, float* counts,
                    float* sums, float* newc, hipStream_t stream) {
  const int B = 1024, K = 200, Kp = 208;
  long rows = (long)B * C;
  qk_knn_normalize<<<dim3((unsigned)(rows / 8)), 256, 0, stream>>>(h, xn, (int)rows, D, DP);
  long ctot = (long)C * Kp * DP;
  qk_cast_centers<<<ew_grid(ctot), 256, 0, stream>>>(c, cent, C, K, Kp, D, DP);
  (void)hipMemsetAsync(counts, 0, (size_t)C * K * sizeof(float), stream);
  int waves = C * (B / 16);
  qk_knn_dist_argmax<DP><<<dim3((unsigned)(waves / 4)), 128, 0, stream>>>(xn, cent, idx, counts, B, C, K, Kp);
  (void)hipMemsetAsync(sums, 0, (size_t)C * K * D * sizeof(float), stream);
  long atot = (long)B * C * D;
  qk_knn_accum_sums<<<ew_grid(atot), 256, 0, stream>>>(xn, idx, sums, B, C, K, D, DP, atot);
  qk_knn_update<<<dim3((unsigned)(C * K / 8)), 256, 0, stream>>>(c, t, counts, sums, newc, C, K, D);
  qk_knn_gather<<<ew_grid(atot), 256, 0, stream>>>(newc, idx, (__bf16*)q_out, B, C, K, D, mode, atot);
}

extern "C" void kernel_launch(void* const* d_in, const int* in_sizes, int n_in,
                              void* d_out, int out_size, void* d_ws, size_t ws_size,
                              hipStream_t stream) {
  (void)in_sizes; (void)n_in; (void)out_size; (void)ws_size;
  const float* x    = (const float*)d_in[0];
  const float* w1   = (const float*)d_in[1];
  const float* b1   = (const float*)d_in[2];
  const float* w2   = (const float*)d_in[3];
  const float* b2   = (const float*)d_in[4];
  const float* w3   = (const float*)d_in[5];
  const float* b3   = (const float*)d_in[6];
  const float* fw1  = (const float*)d_in[7];
  const float* fb1  = (const float*)d_in[8];
  const float* fw2  = (const float*)d_in[9];
  const float* fb2  = (const float*)d_in[10];
  const float* c0   = (const float*)d_in[11];
  const float* c1   = (const float*)d_in[12];
  const float* c2   = (const float*)d_in[13];
  const int*   t0   = (const int*)d_in[14];
  const int*   t1   = (const int*)d_in[15];
  const int*   t2   = (const int*)d_in[16];

  char* ws = (char*)d_ws;
  float*  act    = (float*)(ws + OFF_ACT);
  __bf16* xn     = (__bf16*)(ws + OFF_XN);
  __bf16* cent   = (__bf16*)(ws + OFF_CENT);
  int*    idx    = (int*)(ws + OFF_IDX);
  float*  counts = (float*)(ws + OFF_COUNTS);
  float*  sums   = (float*)(ws + OFF_SUMS);
  float*  newc   = (float*)(ws + OFF_NEWC);
  void*   qbuf   = (void*)(ws + OFF_QBUF);
  __bf16* wpack  = (__bf16*)(ws + OFF_WPACK);
  __bf16* fc1wb  = (__bf16*)(ws + OFF_FC1W);
  __bf16* fc2wb  = (__bf16*)(ws + OFF_FC2W);
  float*  fc1o   = (float*)(ws + OFF_FC1O);
  __bf16* fc1ob  = (__bf16*)(ws + OFF_FC1OB);

  const int B = 1024;

  // stage 1: conv1 + relu + pool -> act (B,32,196) NCHW f32
  qk_conv1_relu_pool<<<ew_grid((long)B * 32 * 196), 256, 0, stream>>>(x, w1, b1, act, B * 32 * 196);

  // stage 2: knn layer 0 -> q0 NHWC bf16 in qbuf
  run_knn<224>(act, c0, t0, 32, 196, qbuf, 0, xn, cent, idx, counts, sums, newc, stream);

  // stage 3: conv2 (WMMA taps) -> act (B,32,196) NCHW f32
  qk_pack_conv_w<<<ew_grid(25L * 32 * 32), 256, 0, stream>>>(w2, wpack, 32, 25 * 32 * 32);
  {
    int waves = ((B * 196) / 16) * (32 / 16);   // 25088
    qk_conv_taps<<<dim3((unsigned)(waves / 4)), 128, 0, stream>>>((const __bf16*)qbuf, wpack, b2, act, B, 32);
  }

  // stage 4: knn layer 1 -> q1 NHWC bf16 in qbuf
  run_knn<224>(act, c1, t1, 32, 196, qbuf, 0, xn, cent, idx, counts, sums, newc, stream);

  // stage 5: conv3 (WMMA taps) -> act (B,64,196) NCHW f32
  qk_pack_conv_w<<<ew_grid(25L * 64 * 32), 256, 0, stream>>>(w3, wpack, 64, 25 * 64 * 32);
  {
    int waves = ((B * 196) / 16) * (64 / 16);   // 50176
    qk_conv_taps<<<dim3((unsigned)(waves / 4)), 128, 0, stream>>>((const __bf16*)qbuf, wpack, b3, act, B, 64);
  }

  // stage 6: maxpool -> h3p f32 (B,64,49) in qbuf
  qk_maxpool2<<<ew_grid((long)B * 64 * 49), 256, 0, stream>>>(act, (float*)qbuf, B * 64 * 49);

  // stage 7: knn layer 2 -> fcin bf16 (B,3136) in qbuf (NCHW-flat, overwrites h3p after use)
  run_knn<64>((const float*)qbuf, c2, t2, 64, 49, qbuf, 1, xn, cent, idx, counts, sums, newc, stream);

  // stage 8: fc weights -> bf16 (fc2 padded 10 -> 16 rows, zeros)
  qk_cast_f32_bf16<<<ew_grid(1024L * 3136), 256, 0, stream>>>(fw1, fc1wb, 1024L * 3136);
  (void)hipMemsetAsync(fc2wb, 0, (size_t)16 * 1024 * sizeof(__bf16), stream);
  qk_cast_f32_bf16<<<ew_grid(10L * 1024), 256, 0, stream>>>(fw2, fc2wb, 10L * 1024);

  // stage 9: fc1 (M=1024,N=1024,K=3136) + bias + relu, mirrored to bf16
  {
    int blocks = (1024 / 64) * (1024 / 16);     // 4 M-tiles per block, 1024 blocks
    qk_gemm_nt<<<dim3((unsigned)blocks), 128, 0, stream>>>(
        (const __bf16*)qbuf, 3136, fc1wb, 3136, fc1o, 1024,
        fb1, 1, fc1ob, 1024, 1024, 1024, 1024, 3136);
  }

  // stage 10: fc2 (M=1024,N=10 pad 16,K=1024) + bias -> d_out f32
  {
    int blocks = (1024 / 64) * (16 / 16);       // 16 blocks
    qk_gemm_nt<<<dim3((unsigned)blocks), 128, 0, stream>>>(
        fc1ob, 1024, fc2wb, 1024, (float*)d_out, 10,
        fb2, 0, (__bf16*)nullptr, 0, 1024, 10, 16, 1024);
  }
}
